// MaskedSelfAttention_73443940761823
// MI455X (gfx1250) — matrix operand order repaired and checked by reading, not compiled, not run on hardware
//
#include <hip/hip_runtime.h>
#include <hip/hip_bf16.h>
#include <math.h>

// Problem constants (reference: B=4, T=2048, C=1024)
#define NB 4
#define TB 2048
#define CB 1024

typedef __bf16 bf16;
typedef __attribute__((ext_vector_type(16))) __bf16 v16bf;
typedef __attribute__((ext_vector_type(8)))  float  v8f;

// ---------------- helpers ----------------

__device__ __forceinline__ unsigned short f2bf_rne(float f) {
  union { float f; unsigned u; } x; x.f = f;
  unsigned r = x.u + 0x7FFFu + ((x.u >> 16) & 1u);   // round-to-nearest-even
  return (unsigned short)(r >> 16);
}
__device__ __forceinline__ unsigned pack2(float a, float b) {
  return (unsigned)f2bf_rne(a) | ((unsigned)f2bf_rne(b) << 16);
}
__device__ __forceinline__ v16bf frag_cast(uint4 lo, uint4 hi) {
  union { uint4 u[2]; v16bf v; } c; c.u[0] = lo; c.u[1] = hi; return c.v;
}

// ---- gfx1250 async global->LDS DMA (ASYNCcnt path, ISA 08_async_tensor §4) ----
// GV addressing: LDS byte address in VDST vgpr, 64-bit global address in VADDR.
__device__ __forceinline__ void async_b128(void* lds_dst, const void* gsrc) {
  unsigned loff = (unsigned)(size_t)lds_dst;  // low 32 bits of generic LDS ptr = LDS offset
  asm volatile("global_load_async_to_lds_b128 %0, %1, off"
               :: "v"(loff), "v"(gsrc) : "memory");
}
__device__ __forceinline__ void wait_async0() {
  asm volatile("s_wait_asynccnt 0x0" ::: "memory");
}

// Stage a 128x32 bf16 tile into LDS via async DMA: 2 x b128 per thread.
__device__ __forceinline__ void stage_async128(unsigned* lds, const bf16* src,
                                               int ldsrc, int tid) {
  int seg = tid & 3, r = tid >> 2;
  async_b128(lds + r * 16 + seg * 4, src + (size_t)r * ldsrc + seg * 8);
  r += 64;
  async_b128(lds + r * 16 + seg * 4, src + (size_t)r * ldsrc + seg * 8);
}

// Synchronous staging with fp32 -> bf16 conversion (nrows x 32 tile).
__device__ __forceinline__ void stage_f32(unsigned* lds, const float* src,
                                          int ldsrc, int nrows, int tid) {
  int seg = tid & 3;
  for (int r = tid >> 2; r < nrows; r += 64) {
    const float4* p = (const float4*)(src + (size_t)r * ldsrc + seg * 8);
    float4 f0 = p[0], f1 = p[1];
    unsigned* d = lds + r * 16 + seg * 4;
    d[0] = pack2(f0.x, f0.y); d[1] = pack2(f0.z, f0.w);
    d[2] = pack2(f1.x, f1.y); d[3] = pack2(f1.z, f1.w);
  }
}
// Synchronous bf16 staging (nrows x 32 tile).
__device__ __forceinline__ void stage_bf16(unsigned* lds, const bf16* src,
                                           int ldsrc, int nrows, int tid) {
  int seg = tid & 3;
  for (int r = tid >> 2; r < nrows; r += 64) {
    uint4 q = *(const uint4*)(src + (size_t)r * ldsrc + seg * 8);
    *(uint4*)(lds + r * 16 + seg * 4) = q;
  }
}

// A fragment: 16x32 bf16 tile, row-major in LDS (16 dwords/row).
__device__ __forceinline__ v16bf load_a_frag(const unsigned* lds_a, int mrow, int lane) {
  int hf = lane >> 4, m = lane & 15;
  const uint4* base = (const uint4*)(lds_a + (mrow + m) * 16);
  return frag_cast(base[hf], base[2 + hf]);
}
// B fragment: 32(K)x16(N) bf16 staged N-major in LDS ([n][k], 16 dwords/row).
__device__ __forceinline__ v16bf load_b_frag(const unsigned* lds_b, int ncol, int lane) {
  int hf = lane >> 4, n = lane & 15;
  const uint4* base = (const uint4*)(lds_b + (ncol + n) * 16);
  return frag_cast(base[hf], base[2 + hf]);
}

// One 32-deep k-step for a wave tile of 32(M) x 64(N): 8 WMMAs.
__device__ __forceinline__ void mma_step(const unsigned* la, const unsigned* lb,
                                         int wm, int wn, int lane, v8f acc[2][4]) {
  v16bf afr[2], bfr[4];
#pragma unroll
  for (int i = 0; i < 2; ++i) afr[i] = load_a_frag(la, wm * 32 + i * 16, lane);
#pragma unroll
  for (int j = 0; j < 4; ++j) bfr[j] = load_b_frag(lb, wn * 64 + j * 16, lane);
#pragma unroll
  for (int i = 0; i < 2; ++i)
#pragma unroll
    for (int j = 0; j < 4; ++j)
      acc[i][j] = __builtin_amdgcn_wmma_f32_16x16x32_bf16(
          false, afr[i], false, bfr[j], (short)0, acc[i][j], false, false);
}

// ---------------- GEMM for projections / output: C = A @ W^T (+bias) ----------------
// WG tile 128x128, 8 waves (4 over M x 2 over N), wave tile 32x64.
// fp32 sources need convert-in-registers, so staging is synchronous here.

template <bool A_F32, bool OUT_BF16, bool TRANS_OUT>
__global__ __launch_bounds__(256) void proj_gemm(
    const void* __restrict__ Ap, const float* __restrict__ W,
    const float* __restrict__ bias, void* __restrict__ Out,
    int M, int N, int K) {
  __shared__ __align__(16) unsigned lds_a[128 * 16];
  __shared__ __align__(16) unsigned lds_b[128 * 16];
  int tid = threadIdx.x, lane = tid & 31, wave = tid >> 5;
  int wm = wave & 3, wn = wave >> 2;
  int row0 = blockIdx.y * 128, col0 = blockIdx.x * 128;

  v8f acc[2][4] = {};
  for (int k0 = 0; k0 < K; k0 += 32) {
    __syncthreads();
    if (A_F32)
      stage_f32(lds_a, (const float*)Ap + (size_t)row0 * K + k0, K, 128, tid);
    else
      stage_bf16(lds_a, (const bf16*)Ap + (size_t)row0 * K + k0, K, 128, tid);
    stage_f32(lds_b, W + (size_t)col0 * K + k0, K, 128, tid);
    __syncthreads();
    mma_step(lds_a, lds_b, wm, wn, lane, acc);
  }
  int hf = lane >> 4, nn = lane & 15;
#pragma unroll
  for (int i = 0; i < 2; ++i)
#pragma unroll
    for (int j = 0; j < 4; ++j) {
      int colb = col0 + wn * 64 + j * 16 + nn;
      float bv = bias ? bias[colb] : 0.0f;
#pragma unroll
      for (int r = 0; r < 8; ++r) {
        int m = row0 + wm * 32 + i * 16 + 8 * hf + r;
        float v = acc[i][j][r] + bv;
        if (OUT_BF16) {
          if (TRANS_OUT) {  // store (batch, N, T): used for V^T
            int bb = m / TB, tt = m % TB;
            ((bf16*)Out)[((size_t)bb * N + colb) * TB + tt] = (bf16)v;
          } else {
            ((bf16*)Out)[(size_t)m * N + colb] = (bf16)v;
          }
        } else {
          ((float*)Out)[(size_t)m * N + colb] = v;
        }
      }
    }
}

// ---------------- Scores: S = scale * Q @ K^T (batched, causal tile-skip) ----------------
// bf16 sources: async DMA staging, double-buffered LDS.

__global__ __launch_bounds__(256) void scores_kernel(
    const bf16* __restrict__ Q, const bf16* __restrict__ Kmat,
    float* __restrict__ S, float scale) {
  int b = blockIdx.z;
  int t0 = blockIdx.y * 128;
  int s0 = blockIdx.x * 128;
  if (s0 > t0 + 127) return;  // tile entirely above the diagonal: never read
  __shared__ __align__(16) unsigned lds_a[2][128 * 16];
  __shared__ __align__(16) unsigned lds_b[2][128 * 16];
  const bf16* Qb = Q + (size_t)b * TB * CB + (size_t)t0 * CB;
  const bf16* Kb = Kmat + (size_t)b * TB * CB + (size_t)s0 * CB;
  float* Sb = S + (size_t)b * TB * TB;
  int tid = threadIdx.x, lane = tid & 31, wave = tid >> 5;
  int wm = wave & 3, wn = wave >> 2;

  const int nk = CB / 32;
  stage_async128(lds_a[0], Qb, CB, tid);
  stage_async128(lds_b[0], Kb, CB, tid);

  v8f acc[2][4] = {};
  for (int kk = 0; kk < nk; ++kk) {
    int cur = kk & 1;
    wait_async0();
    __syncthreads();                       // tile kk visible to all waves
    if (kk + 1 < nk) {                     // prefetch tile kk+1 into other buffer
      stage_async128(lds_a[cur ^ 1], Qb + (kk + 1) * 32, CB, tid);
      stage_async128(lds_b[cur ^ 1], Kb + (kk + 1) * 32, CB, tid);
    }
    mma_step(lds_a[cur], lds_b[cur], wm, wn, lane, acc);
    __syncthreads();                       // everyone done reading buf cur
  }

  int hf = lane >> 4, nn = lane & 15;
#pragma unroll
  for (int i = 0; i < 2; ++i)
#pragma unroll
    for (int j = 0; j < 4; ++j)
#pragma unroll
      for (int r = 0; r < 8; ++r) {
        int t = t0 + wm * 32 + i * 16 + 8 * hf + r;
        int s = s0 + wn * 64 + j * 16 + nn;
        Sb[(size_t)t * TB + s] = acc[i][j][r] * scale;  // s>t entries never read
      }
}

// ---------------- Row softmax: fp32 scores -> bf16 probs (zeros past diagonal) ----------------

__global__ __launch_bounds__(256) void softmax_kernel(
    const float* __restrict__ S, bf16* __restrict__ P) {
  int rid = blockIdx.x;                 // 0 .. NB*TB-1
  int b = rid >> 11, t = rid & (TB - 1);
  const float* row = S + ((size_t)b * TB + t) * TB;
  bf16* prow = P + ((size_t)b * TB + t) * TB;
  int tid = threadIdx.x;
  int nvalid = t + 1;
  __shared__ float red[256];

  float mx = -3.0e38f;
  for (int j = tid; j < nvalid; j += 256) mx = fmaxf(mx, row[j]);
  red[tid] = mx; __syncthreads();
  for (int s = 128; s > 0; s >>= 1) {
    if (tid < s) red[tid] = fmaxf(red[tid], red[tid + s]);
    __syncthreads();
  }
  mx = red[0]; __syncthreads();

  float sum = 0.0f;
  for (int j = tid; j < nvalid; j += 256) sum += __expf(row[j] - mx);
  red[tid] = sum; __syncthreads();
  for (int s = 128; s > 0; s >>= 1) {
    if (tid < s) red[tid] += red[tid + s];
    __syncthreads();
  }
  float inv = 1.0f / red[0];

  for (int j = tid; j < TB; j += 256) {
    float p = (j < nvalid) ? __expf(row[j] - mx) * inv : 0.0f;
    prow[j] = (bf16)p;
  }
}

// ---------------- O = P @ V  (P: T x T bf16, V^T: C x T bf16, causal k-limit) ----------------

__global__ __launch_bounds__(256) void pv_kernel(
    const bf16* __restrict__ P, const bf16* __restrict__ Vt,
    bf16* __restrict__ O) {
  int b = blockIdx.z;
  int t0 = blockIdx.y * 128;
  int c0 = blockIdx.x * 128;
  __shared__ __align__(16) unsigned lds_a[2][128 * 16];
  __shared__ __align__(16) unsigned lds_b[2][128 * 16];
  const bf16* Pb = P + (size_t)b * TB * TB + (size_t)t0 * TB;
  const bf16* Vb = Vt + (size_t)b * CB * TB + (size_t)c0 * TB;
  int tid = threadIdx.x, lane = tid & 31, wave = tid >> 5;
  int wm = wave & 3, wn = wave >> 2;

  const int nk = (t0 + 128) / 32;        // probs are zero for s > t, rest skipped
  stage_async128(lds_a[0], Pb, TB, tid);
  stage_async128(lds_b[0], Vb, TB, tid);

  v8f acc[2][4] = {};
  for (int kk = 0; kk < nk; ++kk) {
    int cur = kk & 1;
    wait_async0();
    __syncthreads();
    if (kk + 1 < nk) {
      stage_async128(lds_a[cur ^ 1], Pb + (kk + 1) * 32, TB, tid);
      stage_async128(lds_b[cur ^ 1], Vb + (kk + 1) * 32, TB, tid);
    }
    mma_step(lds_a[cur], lds_b[cur], wm, wn, lane, acc);
    __syncthreads();
  }

  int hf = lane >> 4, nn = lane & 15;
#pragma unroll
  for (int i = 0; i < 2; ++i)
#pragma unroll
    for (int j = 0; j < 4; ++j)
#pragma unroll
      for (int r = 0; r < 8; ++r) {
        int t = t0 + wm * 32 + i * 16 + 8 * hf + r;
        int c = c0 + wn * 64 + j * 16 + nn;
        O[((size_t)b * TB + t) * CB + c] = (bf16)acc[i][j][r];
      }
}

// ---------------- launch ----------------

extern "C" void kernel_launch(void* const* d_in, const int* in_sizes, int n_in,
                              void* d_out, int out_size, void* d_ws, size_t ws_size,
                              hipStream_t stream) {
  const float* x  = (const float*)d_in[0];
  const float* Wq = (const float*)d_in[1];
  const float* bq = (const float*)d_in[2];
  const float* Wk = (const float*)d_in[3];
  const float* bk = (const float*)d_in[4];
  const float* Wv = (const float*)d_in[5];
  const float* bv = (const float*)d_in[6];
  const float* Wo = (const float*)d_in[7];
  const float* bo = (const float*)d_in[8];
  float* out = (float*)d_out;

  // Workspace: Q,K,V^T,O bf16 (4 x 16 MB) + P bf16 (32 MB) + S fp32 (64 MB) = 160 MB
  char* ws = (char*)d_ws;
  const size_t btc = (size_t)NB * TB * CB;
  bf16* Qb = (bf16*)ws;  ws += btc * 2;
  bf16* Kb = (bf16*)ws;  ws += btc * 2;
  bf16* Vt = (bf16*)ws;  ws += btc * 2;
  bf16* Ob = (bf16*)ws;  ws += btc * 2;
  bf16* Pp = (bf16*)ws;  ws += (size_t)NB * TB * TB * 2;
  float* S = (float*)ws;

  const int M = NB * TB;  // 8192 rows, batch folded into M for projections
  dim3 blk(256);
  dim3 gp(CB / 128, M / 128);

  // Q, K, V projections (V stored transposed for the P@V GEMM)
  proj_gemm<true, true, false><<<gp, blk, 0, stream>>>(x, Wq, bq, Qb, M, CB, CB);
  proj_gemm<true, true, false><<<gp, blk, 0, stream>>>(x, Wk, bk, Kb, M, CB, CB);
  proj_gemm<true, true, true ><<<gp, blk, 0, stream>>>(x, Wv, bv, Vt, M, CB, CB);

  // S = scale * Q K^T  (causal tiles skipped; async-DMA double-buffered)
  dim3 gs(TB / 128, TB / 128, NB);
  scores_kernel<<<gs, blk, 0, stream>>>(Qb, Kb, S, 1.0f / 32.0f);

  // row softmax -> bf16 probs
  softmax_kernel<<<dim3(NB * TB), blk, 0, stream>>>(S, Pp);

  // O = P @ V  (async-DMA double-buffered)
  dim3 gv(CB / 128, TB / 128, NB);
  pv_kernel<<<gv, blk, 0, stream>>>(Pp, Vt, Ob);

  // out = O @ Wo^T + bo   (fp32 output)
  proj_gemm<false, false, false><<<gp, blk, 0, stream>>>(Ob, Wo, bo, out, M, CB, CB);
}
